// SceneEncoder_2250562863674
// MI455X (gfx1250) — matrix operand order repaired and checked by reading, not compiled, not run on hardware
//
#include <hip/hip_runtime.h>
#include <hip/hip_bf16.h>
#include <math.h>

// ---------------- constants ----------------
#define BB 2
#define NA 64
#define NL 256
#define NN 320          // tokens per batch
#define DD 128
#define HH 8
#define HD 16
#define LL 3
#define PAIRS (NN*NN)   // 102400 pairs per batch

typedef __attribute__((ext_vector_type(16))) _Float16 v16h;
typedef __attribute__((ext_vector_type(8)))  float    v8f;

// Fragment layouts (wave32, v_wmma_f32_16x16x32_f16), from CDNA5 ISA 7.12.2:
//  A (16x32 MxK):  lane = (m&15) + 16*kh,  element j -> k = 16*kh + j (code's order, not ISA table)
//  B (32x16 KxN):  lane = (n&15) + 16*(k>>4), element j -> k = (k&~15) + j (code's order, same as A)
//  C/D (16x16 f32): lane = n + 16*(m>=8), vgpr r -> m = r + 8*(lane>=16)
// We stage LDS tiles PRE-SWIZZLED so each lane's 16 f16 elements are contiguous:
// fragment load = one v16h LDS load (2x ds_load_b128), no per-element gathers.

// ---------------- kernel 1: position features ----------------
__global__ void build_pf_kernel(const float* __restrict__ xpos,
                                const float* __restrict__ xang,
                                const float* __restrict__ lpos,
                                float* __restrict__ pf) {
  int idx = blockIdx.x * blockDim.x + threadIdx.x; // b*NN + n
  if (idx >= BB * NN) return;
  int b = idx / NN, n = idx % NN;
  float px, py, ang;
  if (n < NA) {
    size_t base = ((size_t)(b * NA + n) * 50 + 49) * 2;
    px = xpos[base]; py = xpos[base + 1];
    ang = xang[(size_t)(b * NA + n) * 50 + 49];
  } else {
    int l = n - NA;
    size_t base = (size_t)(b * NL + l) * 20 * 2;
    float x0 = lpos[base], y0 = lpos[base + 1];
    float x1 = lpos[base + 2], y1 = lpos[base + 3];
    px = x0; py = y0;
    ang = atan2f(y1 - y0, x1 - x0);
  }
  pf[(size_t)idx * 3 + 0] = px;
  pf[(size_t)idx * 3 + 1] = py;
  pf[(size_t)idx * 3 + 2] = ang;
}

// ---------------- kernel 2: fused pair MLP embedding (pos & neg) ----------------
// grid: BB * (PAIRS/64) blocks, 128 threads (4 waves). Each block: 64 pairs.
// LDS tiles stored in swizzled fragment layout.
__global__ void pair_embed_kernel(const float* __restrict__ pf,
                                  const float* __restrict__ pw1, const float* __restrict__ pb1,
                                  const float* __restrict__ pw2, const float* __restrict__ pb2,
                                  const float* __restrict__ nw1, const float* __restrict__ nb1,
                                  const float* __restrict__ nw2, const float* __restrict__ nb2,
                                  _Float16* __restrict__ rp, _Float16* __restrict__ rpn) {
  __shared__ float relf[64][4];
  // hAsw[wave][kt][lane][j]  : 4*4*32*16 f16 = 16 KB
  __shared__ alignas(32) _Float16 hAsw[4 * 4 * 32 * 16];
  // wBsw[kt][ct][lane][j]    : 4*8*32*16 f16 = 32 KB
  __shared__ alignas(32) _Float16 wBsw[4 * 8 * 32 * 16];

  int t = threadIdx.x;
  int b = blockIdx.x / (PAIRS / 64);
  int tile = blockIdx.x % (PAIRS / 64);
  int base = tile * 64;

  if (t < 64) {
    int flat = base + t;
    int n = flat / NN, m = flat % NN;
    const float* pn = pf + (size_t)(b * NN + n) * 3;
    const float* pm = pf + (size_t)(b * NN + m) * 3;
    float dx = pn[0] - pm[0], dy = pn[1] - pm[1];
    relf[t][0] = dx;
    relf[t][1] = dy;
    relf[t][2] = sqrtf(dx * dx + dy * dy);
    relf[t][3] = pn[2] - pm[2];
  }
  __syncthreads();

  int wave = t >> 5, lane = t & 31;

  for (int pass = 0; pass < 2; ++pass) {
    const float* w1 = pass ? nw1 : pw1;
    const float* b1 = pass ? nb1 : pb1;
    const float* w2 = pass ? nw2 : pw2;
    const float* b2 = pass ? nb2 : pb2;
    _Float16* out = pass ? rpn : rp;

    // hidden = relu(rel @ w1 + b1) -> swizzled A tile. Thread t: pair p = t/2,
    // k-range = (t&1)*64..+64, written as 8 destination-contiguous runs of 8.
    {
      int p = t >> 1;
      int wv = p >> 4, mm = p & 15;
      int k0h = (t & 1) * 64;
      float r0 = relf[p][0], r1 = relf[p][1], r2 = relf[p][2], r3 = relf[p][3];
      if (pass) { r0 = -r0; r1 = -r1; r3 = -r3; }
#pragma unroll
      for (int g = 0; g < 8; ++g) {
        int kbase = k0h + g * 8;
        int kt = kbase >> 5;
        int kk0 = kbase & 31;
        int grp = kk0 >> 3;
        int kh = grp >> 1;
        int j0 = (grp & 1) << 3;
        _Float16* dst = &hAsw[(((wv << 2) + kt) * 32 + mm + (kh << 4)) * 16 + j0];
#pragma unroll
        for (int q = 0; q < 8; ++q) {
          int k = kbase + q;
          float hv = b1[k] + r0 * w1[k] + r1 * w1[128 + k] + r2 * w1[256 + k] + r3 * w1[384 + k];
          dst[q] = (_Float16)fmaxf(hv, 0.f);
        }
      }
    }
    // w2 (128x128) -> swizzled B tiles. 1024 destination runs of 16; 8 per thread.
#pragma unroll
    for (int i = 0; i < 8; ++i) {
      int r = t + 128 * i;
      int lane5 = r & 31;
      int ct = (r >> 5) & 7;
      int kt = r >> 8;
      int n = ct * 16 + (lane5 & 15);
      int krow0 = kt * 32 + ((lane5 >> 4) << 4);
      _Float16* dst = &wBsw[((kt * 8 + ct) * 32 + lane5) * 16];
#pragma unroll
      for (int q = 0; q < 16; ++q)
        dst[q] = (_Float16)w2[(size_t)(krow0 + q) * 128 + n];
    }
    __syncthreads();

    const v16h* Af = (const v16h*)hAsw;
    const v16h* Bf = (const v16h*)wBsw;
    int row0 = wave * 16;
    for (int ct = 0; ct < 8; ++ct) {
      v8f acc = {};
#pragma unroll
      for (int kt = 0; kt < 4; ++kt) {
        v16h a = Af[(wave * 4 + kt) * 32 + lane];
        v16h bf = Bf[(kt * 8 + ct) * 32 + lane];
        acc = __builtin_amdgcn_wmma_f32_16x16x32_f16(false, a, false, bf, (short)0, acc,
                                                     false, false);
      }
      int col = ct * 16 + (lane & 15);
      float bias = b2[col];
#pragma unroll
      for (int r = 0; r < 8; ++r) {
        int mr = row0 + r + ((lane >> 4) << 3);
        size_t off = ((size_t)b * PAIRS + base + mr) * (size_t)DD + col;
        out[off] = (_Float16)(acc[r] + bias);
      }
    }
    __syncthreads();
  }
}

// ---------------- kernel 3: init x = concat(agent, lane) ----------------
__global__ void init_x_kernel(const float* __restrict__ agent,
                              const float* __restrict__ lane,
                              float* __restrict__ x) {
  int row = blockIdx.x, t = threadIdx.x;
  int b = row / NN, n = row % NN;
  float v = (n < NA) ? agent[(size_t)(b * NA + n) * DD + t]
                     : lane[(size_t)(b * NL + (n - NA)) * DD + t];
  x[(size_t)row * DD + t] = v;
}

// ---------------- kernel 4: layernorm (1 row / block, 128 threads) ----------------
__global__ void ln_kernel(const float* __restrict__ x, const float* __restrict__ g,
                          const float* __restrict__ bb, float* __restrict__ out) {
  __shared__ float red[128];
  int row = blockIdx.x, t = threadIdx.x;
  float v = x[(size_t)row * DD + t];
  red[t] = v;
  __syncthreads();
  for (int s = 64; s > 0; s >>= 1) { if (t < s) red[t] += red[t + s]; __syncthreads(); }
  float mu = red[0] * (1.0f / DD);
  __syncthreads();
  float d = v - mu;
  red[t] = d * d;
  __syncthreads();
  for (int s = 64; s > 0; s >>= 1) { if (t < s) red[t] += red[t + s]; __syncthreads(); }
  float var = red[0] * (1.0f / DD);
  out[(size_t)row * DD + t] = d * rsqrtf(var + 1e-5f) * g[t] + bb[t];
}

// ---------------- kernel 5: generic WMMA GEMM (swizzled LDS tiles) ----------------
// out[M,Nout] = X[M,K] @ W[K,Nout] + bias ; mode 0=store, 1=relu store, 2=residual add
// grid (M/64, Nout/64), 128 threads. M%64==0, K%32==0, Nout%64==0.
__global__ void gemm_kernel(const float* __restrict__ X, const float* __restrict__ W,
                            const float* __restrict__ bias, float* __restrict__ out,
                            int M, int K, int Nout, int mode) {
  __shared__ alignas(32) _Float16 Asw[4 * 32 * 16]; // [wave][lane][j]  4 KB
  __shared__ alignas(32) _Float16 Bsw[4 * 32 * 16]; // [ct][lane][j]    4 KB
  int t = threadIdx.x;
  int m0 = blockIdx.x * 64;
  int n0 = blockIdx.y * 64;
  int wave = t >> 5, lane = t & 31;
  v8f acc[4] = {};

  for (int k0 = 0; k0 < K; k0 += 32) {
    // A staging: 256 runs of 8, 2 per thread; source reads contiguous f32.
#pragma unroll
    for (int i = 0; i < 2; ++i) {
      int r = t + 128 * i;
      int m = r >> 2, g = r & 3;
      int kh = g >> 1, j0 = (g & 1) << 3;
      int wv = m >> 4;
      int lane5 = (m & 15) + (kh << 4);
      const float* src = &X[(size_t)(m0 + m) * K + k0 + g * 8];
      _Float16* dst = &Asw[(wv * 32 + lane5) * 16 + j0];
#pragma unroll
      for (int q = 0; q < 8; ++q) dst[q] = (_Float16)src[q];
    }
    // B staging: 128 runs of 16, 1 per thread; reads coalesced across wave per q.
    {
      int lane5 = t & 31;
      int ct = t >> 5;
      int n = n0 + ct * 16 + (lane5 & 15);
      int krow0 = k0 + ((lane5 >> 4) << 4);
      _Float16* dst = &Bsw[(ct * 32 + lane5) * 16];
#pragma unroll
      for (int q = 0; q < 16; ++q)
        dst[q] = (_Float16)W[(size_t)(krow0 + q) * Nout + n];
    }
    __syncthreads();
    v16h a = ((const v16h*)Asw)[wave * 32 + lane];
#pragma unroll
    for (int ct = 0; ct < 4; ++ct) {
      v16h bf = ((const v16h*)Bsw)[ct * 32 + lane];
      acc[ct] = __builtin_amdgcn_wmma_f32_16x16x32_f16(false, a, false, bf, (short)0,
                                                       acc[ct], false, false);
    }
    __syncthreads();
  }
#pragma unroll
  for (int ct = 0; ct < 4; ++ct) {
    int col = n0 + ct * 16 + (lane & 15);
    float bv = bias ? bias[col] : 0.f;
#pragma unroll
    for (int r = 0; r < 8; ++r) {
      int row = m0 + wave * 16 + r + ((lane >> 4) << 3);
      float v = acc[ct][r] + bv;
      if (mode == 1) v = fmaxf(v, 0.f);
      size_t off = (size_t)row * Nout + col;
      if (mode == 2) out[off] += v;
      else out[off] = v;
    }
  }
}

// ---------------- kernel 6: fused attention with relative bias ----------------
// grid: BB*HH*NN blocks, 320 threads (thread = key index m).
__global__ void attn_kernel(const float* __restrict__ qkv,
                            const _Float16* __restrict__ rp,
                            const _Float16* __restrict__ rpn,
                            const unsigned char* __restrict__ xmask,
                            const unsigned char* __restrict__ lmask,
                            float* __restrict__ ctx) {
  __shared__ float qs[16];
  __shared__ float sc[NN];
  __shared__ float red[64];
  __shared__ float part[NN];
  int blk = blockIdx.x;
  int b = blk / (HH * NN);
  int rem = blk % (HH * NN);
  int hh = rem / NN;
  int n = rem % NN;
  int t = threadIdx.x; // = m

  if (t < 16) qs[t] = qkv[(size_t)(b * NN + n) * (3 * DD) + hh * HD + t];
  __syncthreads();

  const float* krow = qkv + (size_t)(b * NN + t) * (3 * DD) + DD + hh * HD;
  const _Float16* rprow = rp + (((size_t)(b * NN + n)) * NN + t) * DD + hh * HD;
  float s = 0.f;
#pragma unroll
  for (int d = 0; d < HD; ++d) s += qs[d] * (krow[d] + (float)rprow[d]);
  bool masked = (t < NA) ? (xmask[b * NA + t] != 0) : (lmask[b * NL + (t - NA)] != 0);
  s = masked ? -1e9f : s * 0.25f; // scale = 1/sqrt(16)
  sc[t] = s;
  __syncthreads();

  // max reduction
  if (t < 64) {
    float mx = sc[t];
    for (int j = t + 64; j < NN; j += 64) mx = fmaxf(mx, sc[j]);
    red[t] = mx;
  }
  __syncthreads();
  if (t == 0) {
    float mx = red[0];
    for (int j = 1; j < 64; ++j) mx = fmaxf(mx, red[j]);
    red[0] = mx;
  }
  __syncthreads();
  float mx = red[0];
  __syncthreads();
  float e = __expf(s - mx);
  sc[t] = e;
  __syncthreads();
  if (t < 64) {
    float su = sc[t];
    for (int j = t + 64; j < NN; j += 64) su += sc[j];
    red[t] = su;
  }
  __syncthreads();
  if (t == 0) {
    float su = 0.f;
    for (int j = 0; j < 64; ++j) su += red[j];
    red[0] = su;
  }
  __syncthreads();
  float inv = 1.f / red[0];

  // ctx[d] = sum_m a[m] * (v[m,d] + rpn[n,m,d]) ; a unnormalized in sc, scale at end
  int d = t & 15, chunk = t >> 4; // 20 chunks of 16 keys
  const float* vbase = qkv + (size_t)(b * NN) * (3 * DD) + 2 * DD + hh * HD + d;
  const _Float16* rpnbase = rpn + ((size_t)(b * NN + n)) * NN * DD + hh * HD + d;
  float p = 0.f;
#pragma unroll
  for (int j = 0; j < 16; ++j) {
    int mm = chunk * 16 + j;
    p += sc[mm] * (vbase[(size_t)mm * (3 * DD)] + (float)rpnbase[(size_t)mm * DD]);
  }
  part[t] = p;
  __syncthreads();
  if (t < 16) {
    float tot = 0.f;
    for (int c = 0; c < 20; ++c) tot += part[c * 16 + t];
    ctx[(size_t)(b * NN + n) * DD + hh * HD + t] = tot * inv;
  }
}

// ---------------- host orchestration ----------------
extern "C" void kernel_launch(void* const* d_in, const int* in_sizes, int n_in,
                              void* d_out, int out_size, void* d_ws, size_t ws_size,
                              hipStream_t stream) {
  (void)in_sizes; (void)n_in; (void)out_size; (void)ws_size;
  const float* agent = (const float*)d_in[0];
  const float* lanef = (const float*)d_in[1];
  const float* xpos  = (const float*)d_in[2];
  const float* xang  = (const float*)d_in[3];
  const float* lpos  = (const float*)d_in[4];
  const unsigned char* xmask = (const unsigned char*)d_in[5];
  const unsigned char* lmask = (const unsigned char*)d_in[6];
  const float* pw1 = (const float*)d_in[7];
  const float* pb1 = (const float*)d_in[8];
  const float* pw2 = (const float*)d_in[9];
  const float* pb2 = (const float*)d_in[10];
  const float* nw1 = (const float*)d_in[11];
  const float* nb1 = (const float*)d_in[12];
  const float* nw2 = (const float*)d_in[13];
  const float* nb2 = (const float*)d_in[14];
  const float* qkv_w = (const float*)d_in[15];
  const float* qkv_b = (const float*)d_in[16];
  const float* proj_w = (const float*)d_in[17];
  const float* proj_b = (const float*)d_in[18];
  const float* ln1_g = (const float*)d_in[19];
  const float* ln1_b = (const float*)d_in[20];
  const float* fc1_w = (const float*)d_in[21];
  const float* fc1_b = (const float*)d_in[22];
  const float* fc2_w = (const float*)d_in[23];
  const float* fc2_b = (const float*)d_in[24];
  const float* ln2_g = (const float*)d_in[25];
  const float* ln2_b = (const float*)d_in[26];

  const int M = BB * NN; // 640 rows

  float* ws   = (float*)d_ws;
  float* pf   = ws;                        // 2048 floats reserved
  float* hbuf = ws + 2048;                 // 640*128
  float* qkvb = hbuf + (size_t)M * DD;     // 640*384
  float* ctx  = qkvb + (size_t)M * 3 * DD; // 640*128
  float* ffh  = ctx + (size_t)M * DD;      // 640*512
  _Float16* rp  = (_Float16*)(ffh + (size_t)M * 4 * DD);
  _Float16* rpn = rp + (size_t)BB * PAIRS * DD;

  float* x = (float*)d_out; // activation lives in d_out

  build_pf_kernel<<<(BB * NN + 127) / 128, 128, 0, stream>>>(xpos, xang, lpos, pf);
  pair_embed_kernel<<<BB * (PAIRS / 64), 128, 0, stream>>>(
      pf, pw1, pb1, pw2, pb2, nw1, nb1, nw2, nb2, rp, rpn);
  init_x_kernel<<<M, DD, 0, stream>>>(agent, lanef, x);

  for (int i = 0; i < LL; ++i) {
    ln_kernel<<<M, DD, 0, stream>>>(x, ln1_g + i * DD, ln1_b + i * DD, hbuf);
    gemm_kernel<<<dim3(M / 64, (3 * DD) / 64), 128, 0, stream>>>(
        hbuf, qkv_w + (size_t)i * DD * 3 * DD, qkv_b + (size_t)i * 3 * DD, qkvb,
        M, DD, 3 * DD, 0);
    attn_kernel<<<BB * HH * NN, NN, 0, stream>>>(qkvb, rp, rpn, xmask, lmask, ctx);
    gemm_kernel<<<dim3(M / 64, DD / 64), 128, 0, stream>>>(
        ctx, proj_w + (size_t)i * DD * DD, proj_b + (size_t)i * DD, x,
        M, DD, DD, 2);
    ln_kernel<<<M, DD, 0, stream>>>(x, ln2_g + i * DD, ln2_b + i * DD, hbuf);
    gemm_kernel<<<dim3(M / 64, (4 * DD) / 64), 128, 0, stream>>>(
        hbuf, fc1_w + (size_t)i * DD * 4 * DD, fc1_b + (size_t)i * 4 * DD, ffh,
        M, DD, 4 * DD, 1);
    gemm_kernel<<<dim3(M / 64, DD / 64), 128, 0, stream>>>(
        ffh, fc2_w + (size_t)i * 4 * DD * DD, fc2_b + (size_t)i * DD, x,
        M, 4 * DD, DD, 2);
  }
}
